// GraphAutoencoder_52699248722457
// MI455X (gfx1250) — matrix operand order repaired and checked by reading, not compiled, not run on hardware
//
#include <hip/hip_runtime.h>
#include <hip/hip_bf16.h>

typedef __attribute__((ext_vector_type(16))) __bf16          v16bf;
typedef __attribute__((ext_vector_type(8)))  float           v8f;
typedef __attribute__((ext_vector_type(8)))  unsigned short  u16x8;

#define N_NODES 16384
#define N_EDGES 524288
#define IN_DIM  512
#define HID     64
#define EMB     32

static __device__ __forceinline__ unsigned short f2bf(float f) {
    unsigned int u = __float_as_uint(f);
    u += 0x7FFFu + ((u >> 16) & 1u);          // round-to-nearest-even
    return (unsigned short)(u >> 16);
}

// fast sigmoid: 1 v_exp_f32 + 1 add + 1 v_rcp_f32 (no IEEE divide expansion)
static __device__ __forceinline__ float fast_sigmoid(float x) {
    float e = __builtin_amdgcn_exp2f(x * -1.44269504088896340736f); // exp(-x)
    return __builtin_amdgcn_rcpf(1.0f + e);
}

// ---------------- init: deg=1 (self loop), zero accumulators ----------------
__global__ void k_init(float* __restrict__ deg, float* __restrict__ agg1,
                       float* __restrict__ agg2) {
    int i = blockIdx.x * blockDim.x + threadIdx.x;
    if (i < N_NODES * HID) agg1[i] = 0.0f;
    if (i < N_NODES * EMB) agg2[i] = 0.0f;
    if (i < N_NODES)       deg[i]  = 1.0f;
}

// ---------------- edge prep: int64 -> int32, degree count ----------------
__global__ void k_edges_prep(const long long* __restrict__ ei,
                             int* __restrict__ s32, int* __restrict__ d32,
                             float* __restrict__ deg) {
    int e = blockIdx.x * blockDim.x + threadIdx.x;
    if (e >= N_EDGES) return;
    int s = (int)ei[e];
    int d = (int)ei[N_EDGES + e];
    s32[e] = s;
    d32[e] = d;
    atomicAdd(&deg[d], 1.0f);
}

__global__ void k_dinv(const float* __restrict__ deg, float* __restrict__ dinv) {
    int i = blockIdx.x * blockDim.x + threadIdx.x;
    if (i < N_NODES) dinv[i] = rsqrtf(deg[i]);   // deg >= 1 always (self loop)
}

__global__ void k_norm(const int* __restrict__ s32, const int* __restrict__ d32,
                       const float* __restrict__ dinv, float* __restrict__ nrm) {
    int e = blockIdx.x * blockDim.x + threadIdx.x;
    if (e < N_EDGES) nrm[e] = dinv[s32[e]] * dinv[d32[e]];
}

// ---------------- GEMM1: h0 = x @ W1   [16384x512]x[512x64], fp32 ----------------
__global__ void __launch_bounds__(256) k_gemm1(const float* __restrict__ x,
                                               const float* __restrict__ W1,
                                               float* __restrict__ h0) {
    __shared__ float w[128 * HID];                 // 32 KB K-tile of W1
    const int j   = threadIdx.x & (HID - 1);       // 0..63
    const int r   = threadIdx.x >> 6;              // 0..3
    const int row = blockIdx.x * 4 + r;
    float acc = 0.0f;
    for (int k0 = 0; k0 < IN_DIM; k0 += 128) {
        for (int t = threadIdx.x; t < 128 * HID; t += 256)
            w[t] = W1[(size_t)(k0 + (t >> 6)) * HID + (t & 63)];
        __syncthreads();
        const float* xr = x + (size_t)row * IN_DIM + k0;
        #pragma unroll 8
        for (int k = 0; k < 128; ++k)
            acc = fmaf(xr[k], w[k * HID + j], acc);
        __syncthreads();
    }
    h0[(size_t)row * HID + j] = acc;
}

// ---------------- edge scatter, layer 1 (64 feats/edge) ----------------
__global__ void k_scatter1(const int* __restrict__ s32, const int* __restrict__ d32,
                           const float* __restrict__ nrm, const float* __restrict__ h0,
                           float* __restrict__ agg1) {
    unsigned int t = blockIdx.x * 256u + threadIdx.x;   // < E*64 = 33.5M
    unsigned int e = t >> 6;
    int j = t & 63;
    int s = s32[e], d = d32[e];
    atomicAdd(&agg1[(size_t)d * HID + j], nrm[e] * h0[(size_t)s * HID + j]);
}

// ---------------- finalize layer 1: self-loop + bias + relu (in place) ----------------
__global__ void k_fin1(float* __restrict__ agg1, const float* __restrict__ h0,
                       const float* __restrict__ dinv, const float* __restrict__ b1) {
    unsigned int i = blockIdx.x * 256u + threadIdx.x;   // < N*64
    unsigned int n = i >> 6;
    int j = i & 63;
    float di = dinv[n];
    float v  = agg1[i] + di * di * h0[i] + b1[j];
    agg1[i] = v > 0.0f ? v : 0.0f;                      // relu; agg1 now holds h1
}

// ---------------- GEMM2: z2 = h1 @ W2   [16384x64]x[64x32], fp32 ----------------
__global__ void __launch_bounds__(256) k_gemm2(const float* __restrict__ h1,
                                               const float* __restrict__ W2,
                                               float* __restrict__ z2) {
    __shared__ float w[HID * EMB];                      // 8 KB, full W2
    for (int t = threadIdx.x; t < HID * EMB; t += 256) w[t] = W2[t];
    __syncthreads();
    const int j   = threadIdx.x & 31;
    const int r   = threadIdx.x >> 5;                   // 0..7
    const int row = blockIdx.x * 8 + r;
    const float* hr = h1 + (size_t)row * HID;
    float acc = 0.0f;
    #pragma unroll
    for (int k = 0; k < HID; ++k)
        acc = fmaf(hr[k], w[k * EMB + j], acc);
    z2[(size_t)row * EMB + j] = acc;
}

// ---------------- edge scatter, layer 2 (32 feats/edge) ----------------
__global__ void k_scatter2(const int* __restrict__ s32, const int* __restrict__ d32,
                           const float* __restrict__ nrm, const float* __restrict__ z2,
                           float* __restrict__ agg2) {
    unsigned int t = blockIdx.x * 256u + threadIdx.x;   // < E*32
    unsigned int e = t >> 5;
    int j = t & 31;
    int s = s32[e], d = d32[e];
    atomicAdd(&agg2[(size_t)d * EMB + j], nrm[e] * z2[(size_t)s * EMB + j]);
}

// ---------------- finalize layer 2: z (fp32 out) + bf16 copy for WMMA ----------------
__global__ void k_fin2(const float* __restrict__ agg2, const float* __restrict__ z2,
                       const float* __restrict__ dinv, const float* __restrict__ b2,
                       float* __restrict__ zout, unsigned short* __restrict__ zb) {
    unsigned int i = blockIdx.x * 256u + threadIdx.x;   // < N*32
    unsigned int n = i >> 5;
    int j = i & 31;
    float di = dinv[n];
    float v  = agg2[i] + di * di * z2[i] + b2[j];
    zout[i] = v;
    zb[i]   = f2bf(v);
}

// ---------------- adj_hat = sigmoid(z @ z^T) via v_wmma_f32_16x16x32_bf16 ----------------
// One wave = one 16-row i-tile x four 16-col j-tiles (A fragment reused in regs).
// A layout (16x32 bf16): lane l (m=l&15, hi=l>>4) holds row m,
//   K = 16*hi..16*hi+15 (same k order as B).
// B layout (32x16 bf16): lane l holds column (l&15), K = 16*hi..16*hi+15
//   -> contiguous row (j0 + l&15) of z, offset 16*hi.
// C layout: VGPR r -> row r + 8*hi, col l&15.
// Stores are non-temporal: 1.07 GB written once, never re-read; keep zb in L2.
__global__ void __launch_bounds__(256) k_adj(const unsigned short* __restrict__ zb,
                                             float* __restrict__ out) {
    const int lane   = threadIdx.x & 31;
    const int wave   = (blockIdx.x << 3) | (threadIdx.x >> 5);
    const int itile  = wave >> 8;            // 0..1023
    const int jstrip = wave & 255;           // 0..255
    const int i0 = itile << 4;
    const int j0 = jstrip << 6;
    const int m  = lane & 15;
    const int hi = lane >> 4;

    union Frag { v16bf v; u16x8 u[2]; };

    Frag a;
    const unsigned short* arow = zb + (size_t)(i0 + m) * EMB;
    a.u[0] = *(const u16x8*)(arow + hi * 16);
    a.u[1] = *(const u16x8*)(arow + hi * 16 + 8);

    #pragma unroll
    for (int t = 0; t < 4; ++t) {
        const int jt = j0 + (t << 4);
        Frag b;
        const unsigned short* brow = zb + (size_t)(jt + m) * EMB + hi * 16;
        b.u[0] = *(const u16x8*)(brow);
        b.u[1] = *(const u16x8*)(brow + 8);

        v8f c = {0.f, 0.f, 0.f, 0.f, 0.f, 0.f, 0.f, 0.f};
        c = __builtin_amdgcn_wmma_f32_16x16x32_bf16(
                /*neg_a=*/false, a.v, /*neg_b=*/false, b.v,
                /*c_mod=*/(short)0, c, /*reuse_a=*/false, /*reuse_b=*/false);

        float* orow = out + (size_t)(i0 + hi * 8) * N_NODES + (size_t)jt + m;
        #pragma unroll
        for (int r = 0; r < 8; ++r) {
            __builtin_nontemporal_store(fast_sigmoid(c[r]), orow + (size_t)r * N_NODES);
        }
    }
}

extern "C" void kernel_launch(void* const* d_in, const int* in_sizes, int n_in,
                              void* d_out, int out_size, void* d_ws, size_t ws_size,
                              hipStream_t stream) {
    (void)in_sizes; (void)n_in; (void)out_size; (void)ws_size;

    const float*     x  = (const float*)d_in[0];
    const long long* ei = (const long long*)d_in[1];   // int64 edge_index [2, E]
    const float*     W1 = (const float*)d_in[2];
    const float*     b1 = (const float*)d_in[3];
    const float*     W2 = (const float*)d_in[4];
    const float*     b2 = (const float*)d_in[5];

    float* adj  = (float*)d_out;                                   // [N, N]
    float* zout = adj + (size_t)N_NODES * N_NODES;                 // [N, 32]

    char* ws = (char*)d_ws;                                        // all offsets 64B-aligned
    float* deg  = (float*)ws;  ws += (size_t)N_NODES * 4;
    float* dinv = (float*)ws;  ws += (size_t)N_NODES * 4;
    int*   s32  = (int*)ws;    ws += (size_t)N_EDGES * 4;
    int*   d32  = (int*)ws;    ws += (size_t)N_EDGES * 4;
    float* nrm  = (float*)ws;  ws += (size_t)N_EDGES * 4;
    float* h0   = (float*)ws;  ws += (size_t)N_NODES * HID * 4;
    float* agg1 = (float*)ws;  ws += (size_t)N_NODES * HID * 4;
    float* z2   = (float*)ws;  ws += (size_t)N_NODES * EMB * 4;
    float* agg2 = (float*)ws;  ws += (size_t)N_NODES * EMB * 4;
    unsigned short* zb = (unsigned short*)ws;  // N*EMB bf16 (1 MB, L2-resident)

    k_init      <<<(N_NODES * HID + 255) / 256, 256, 0, stream>>>(deg, agg1, agg2);
    k_edges_prep<<<(N_EDGES + 255) / 256,       256, 0, stream>>>(ei, s32, d32, deg);
    k_dinv      <<<(N_NODES + 255) / 256,       256, 0, stream>>>(deg, dinv);
    k_norm      <<<(N_EDGES + 255) / 256,       256, 0, stream>>>(s32, d32, dinv, nrm);
    k_gemm1     <<<N_NODES / 4,                 256, 0, stream>>>(x, W1, h0);
    k_scatter1  <<<(N_EDGES * HID) / 256,       256, 0, stream>>>(s32, d32, nrm, h0, agg1);
    k_fin1      <<<(N_NODES * HID) / 256,       256, 0, stream>>>(agg1, h0, dinv, b1);
    k_gemm2     <<<N_NODES / 8,                 256, 0, stream>>>(agg1, W2, z2);
    k_scatter2  <<<(N_EDGES * EMB) / 256,       256, 0, stream>>>(s32, d32, nrm, z2, agg2);
    k_fin2      <<<(N_NODES * EMB) / 256,       256, 0, stream>>>(agg2, z2, dinv, b2, zout, zb);
    // 1024 i-tiles x 256 j-strips = 262144 waves, 8 waves/block
    k_adj       <<<32768,                       256, 0, stream>>>(zb, adj);
}
